// MultiHeadedAttention_3435973837163
// MI455X (gfx1250) — matrix operand order repaired and checked by reading, not compiled, not run on hardware
//
#include <hip/hip_runtime.h>
#include <hip/hip_bf16.h>

typedef __bf16 bf16;
typedef __attribute__((ext_vector_type(16))) __bf16 v16bf;
typedef __attribute__((ext_vector_type(8)))  __bf16 v8bf;
typedef __attribute__((ext_vector_type(8)))  float  v8f;
typedef __attribute__((ext_vector_type(4)))  int    v4i;

#define D_MODEL 1024
#define HEADS   16
#define DKH     64
#define BATCH   4
#define SEQ     2048
#define MTOT    (BATCH*SEQ)   /* 8192 */

// ---- async global->LDS path (gfx1250); falls back to reg staging ------------
#if defined(__has_builtin)
#if __has_builtin(__builtin_amdgcn_global_load_async_to_lds_b128) && \
    __has_builtin(__builtin_amdgcn_s_wait_asynccnt)
#define USE_ASYNC_LDS 1
#endif
#endif

__device__ __forceinline__ void copy16_to_lds(const bf16* g, bf16* l) {
#ifdef USE_ASYNC_LDS
  __builtin_amdgcn_global_load_async_to_lds_b128(
      (__attribute__((address_space(1))) v4i*)(v4i*)const_cast<bf16*>(g),
      (__attribute__((address_space(3))) v4i*)(v4i*)l, 0, 0);
#else
  *(v8bf*)l = *(const v8bf*)g;
#endif
}
__device__ __forceinline__ void lds_copy_wait() {
#ifdef USE_ASYNC_LDS
  __builtin_amdgcn_s_wait_asynccnt(0);
#endif
}

union Frag16 { v16bf v; v8bf h[2]; };

__device__ __forceinline__ v8f wmma_bf16(v16bf a, v16bf b, v8f c) {
  return __builtin_amdgcn_wmma_f32_16x16x32_bf16(false, a, false, b, (short)0, c,
                                                 false, false);
}

// ---------------------------------------------------------------- convert ----
__global__ void cvt_f32_bf16(const float* __restrict__ in, bf16* __restrict__ out,
                             int n) {
  int i = blockIdx.x * blockDim.x + threadIdx.x;
  if (i < n) out[i] = (bf16)in[i];
}

// ------------------------------------------------------------------- GEMM ----
// C[M x 1024] = A[M x 1024] (bf16, row-major) @ W^T (W: [1024 x 1024], row n has
// K contiguous) + bias.  Block tile 128x128, 8 waves (4m x 2n), wave tile 32x64.
// K-step 32 (one WMMA), double-buffered LDS with padded stride; async staging.
#define GBM 128
#define GBN 128
#define GKB 32
#define GLD 40   /* padded LDS row stride (elements) */

enum { MODE_F32 = 0, MODE_QK = 1, MODE_VT = 2 };

template <int MODE>
__global__ __launch_bounds__(256)
void gemm_bf16(const bf16* __restrict__ A, const bf16* __restrict__ W,
               const float* __restrict__ bias, void* __restrict__ out) {
  __shared__ bf16 Alds[2][GBM * GLD];
  __shared__ bf16 Blds[2][GBN * GLD];

  const int tid  = threadIdx.x;
  const int lane = tid & 31, w = tid >> 5;
  const int wm = w & 3, wn = w >> 2;      // 4 x 2 wave grid
  const int lm = lane & 15, hi = lane >> 4;

  const int rowBase = blockIdx.y * GBM;
  const int colBase = blockIdx.x * GBN;

  const int ldRow = tid >> 1;             // 0..127
  const int ldSeg = (tid & 1) * 16;       // 0 or 16
  const int K = D_MODEL, nk = K / GKB;    // 32 k-steps

  auto stage = [&](int buf, int kt) {
    const bf16* pa = A + (size_t)(rowBase + ldRow) * K + kt * GKB + ldSeg;
    copy16_to_lds(pa,     &Alds[buf][ldRow * GLD + ldSeg]);
    copy16_to_lds(pa + 8, &Alds[buf][ldRow * GLD + ldSeg + 8]);
    const bf16* pb = W + (size_t)(colBase + ldRow) * K + kt * GKB + ldSeg;
    copy16_to_lds(pb,     &Blds[buf][ldRow * GLD + ldSeg]);
    copy16_to_lds(pb + 8, &Blds[buf][ldRow * GLD + ldSeg + 8]);
  };

  v8f acc[2][4];
#pragma unroll
  for (int i = 0; i < 2; ++i)
#pragma unroll
    for (int j = 0; j < 4; ++j)
#pragma unroll
      for (int r = 0; r < 8; ++r) acc[i][j][r] = 0.f;

  stage(0, 0);
  lds_copy_wait();
  __syncthreads();

  for (int kt = 0; kt < nk; ++kt) {
    const int cur = kt & 1;
    if (kt + 1 < nk) stage(cur ^ 1, kt + 1);  // in flight during compute

    Frag16 fa[2], fb[4];
#pragma unroll
    for (int i = 0; i < 2; ++i) {
      const bf16* p = &Alds[cur][(wm * 32 + i * 16 + lm) * GLD + hi * 8];
      fa[i].h[0] = *(const v8bf*)p;
      fa[i].h[1] = *(const v8bf*)(p + 16);
    }
#pragma unroll
    for (int j = 0; j < 4; ++j) {
      const bf16* p = &Blds[cur][(wn * 64 + j * 16 + lm) * GLD + hi * 8];
      fb[j].h[0] = *(const v8bf*)p;
      fb[j].h[1] = *(const v8bf*)(p + 16);
    }
#pragma unroll
    for (int i = 0; i < 2; ++i)
#pragma unroll
      for (int j = 0; j < 4; ++j)
        acc[i][j] = wmma_bf16(fa[i].v, fb[j].v, acc[i][j]);

    lds_copy_wait();
    __syncthreads();
  }

  // epilogue: C element (VGPR r, lane) -> M = r + 8*hi, N = lm within tile
#pragma unroll
  for (int i = 0; i < 2; ++i) {
#pragma unroll
    for (int j = 0; j < 4; ++j) {
      const int col  = colBase + wn * 64 + j * 16 + lm;
      const float bc = bias[col];
#pragma unroll
      for (int r = 0; r < 8; ++r) {
        const int row = rowBase + wm * 32 + i * 16 + r + 8 * hi;
        const float val = acc[i][j][r] + bc;
        if (MODE == MODE_F32) {
          ((float*)out)[(size_t)row * D_MODEL + col] = val;
        } else {
          const int b = row >> 11, s = row & (SEQ - 1);
          const int hh = col >> 6, dk = col & 63;
          if (MODE == MODE_QK)   // [B,H,S,DK]
            ((bf16*)out)[(((size_t)(b * HEADS + hh)) * SEQ + s) * DKH + dk] =
                (bf16)val;
          else                   // MODE_VT: [B,H,DK,S]
            ((bf16*)out)[(((size_t)(b * HEADS + hh)) * DKH + dk) * SEQ + s] =
                (bf16)val;
        }
      }
    }
  }
}

// -------------------------------------------------------- flash attention ----
// grid (S/128, B*H); block 256 = 8 waves; wave owns 16 q-rows, full d_k=64.
#define NKEY 64
#define ALD  72   /* padded LDS stride */

__global__ __launch_bounds__(256)
void attention_fused(const bf16* __restrict__ Q,   // [B,H,S,DK]
                     const bf16* __restrict__ Kk,  // [B,H,S,DK]
                     const bf16* __restrict__ Vt,  // [B,H,DK,S]
                     const int*  __restrict__ mask,// [B,1,S]
                     bf16* __restrict__ ctx) {     // [B,S,D_MODEL]
  __shared__ bf16 Klds[NKEY * ALD];
  __shared__ bf16 Vlds[DKH * ALD];
  __shared__ bf16 Plds[8][16 * ALD];

  const int tid  = threadIdx.x;
  const int lane = tid & 31, w = tid >> 5;
  const int lm = lane & 15, hi = lane >> 4;
  const int bh = blockIdx.y;
  const int b  = bh >> 4, h = bh & 15;     // H == 16
  const int qblk = blockIdx.x;

  const size_t qkBase = (size_t)bh * SEQ * DKH;
  const size_t vtBase = (size_t)bh * DKH * SEQ;

  // resident Q fragments (A-operand layout: lane holds row lm, k = 8*hi+s, then 16+8*hi+s)
  Frag16 qf[2];
  {
    const int qrow = qblk * 128 + w * 16 + lm;
    const bf16* p = Q + qkBase + (size_t)qrow * DKH;
#pragma unroll
    for (int kc = 0; kc < 2; ++kc) {
      qf[kc].h[0] = *(const v8bf*)(p + kc * 32 + hi * 8);
      qf[kc].h[1] = *(const v8bf*)(p + kc * 32 + 16 + hi * 8);
    }
  }

  v8f acc[4];
#pragma unroll
  for (int j = 0; j < 4; ++j)
#pragma unroll
    for (int r = 0; r < 8; ++r) acc[j][r] = 0.f;

  float mrow[8], lrow[8];
#pragma unroll
  for (int r = 0; r < 8; ++r) { mrow[r] = -1e30f; lrow[r] = 0.f; }

  const int ldRow = tid >> 2;            // 0..63
  const int ldSeg = (tid & 3) * 16;      // 0/16/32/48
  const int nkt = SEQ / NKEY;            // 32 key tiles

  for (int kt = 0; kt < nkt; ++kt) {
    {  // stage K and V^T tiles (async direct-to-LDS when available)
      const bf16* pk = Kk + qkBase + (size_t)(kt * NKEY + ldRow) * DKH + ldSeg;
      copy16_to_lds(pk,     &Klds[ldRow * ALD + ldSeg]);
      copy16_to_lds(pk + 8, &Klds[ldRow * ALD + ldSeg + 8]);
      const bf16* pv = Vt + vtBase + (size_t)ldRow * SEQ + kt * NKEY + ldSeg;
      copy16_to_lds(pv,     &Vlds[ldRow * ALD + ldSeg]);
      copy16_to_lds(pv + 8, &Vlds[ldRow * ALD + ldSeg + 8]);
      if (kt + 1 < nkt)
        __builtin_prefetch(
            Kk + qkBase + (size_t)((kt + 1) * NKEY + ldRow) * DKH + ldSeg, 0, 0);
    }
    lds_copy_wait();
    __syncthreads();

    // S = Q K^T  (16 q-rows x 64 keys per wave)
    v8f s[4];
#pragma unroll
    for (int nt = 0; nt < 4; ++nt)
#pragma unroll
      for (int r = 0; r < 8; ++r) s[nt][r] = 0.f;
#pragma unroll
    for (int nt = 0; nt < 4; ++nt) {
#pragma unroll
      for (int kc = 0; kc < 2; ++kc) {
        Frag16 kb;
        const bf16* p = &Klds[(nt * 16 + lm) * ALD + kc * 32 + hi * 8];
        kb.h[0] = *(const v8bf*)p;
        kb.h[1] = *(const v8bf*)(p + 16);
        s[nt] = wmma_bf16(qf[kc].v, kb.v, s[nt]);
      }
    }

    // mask + 1/sqrt(dk) scale (mask depends only on key column = lane)
#pragma unroll
    for (int nt = 0; nt < 4; ++nt) {
      const int key = kt * NKEY + nt * 16 + lm;
      const int mv  = mask[b * SEQ + key];
#pragma unroll
      for (int r = 0; r < 8; ++r)
        s[nt][r] = mv ? s[nt][r] * 0.125f : -1.0e9f;
    }

    // online softmax stats (reduce across 16-lane groups)
    float rmax[8];
#pragma unroll
    for (int r = 0; r < 8; ++r) {
      float v = fmaxf(fmaxf(s[0][r], s[1][r]), fmaxf(s[2][r], s[3][r]));
      v = fmaxf(v, __shfl_xor(v, 1, 32));
      v = fmaxf(v, __shfl_xor(v, 2, 32));
      v = fmaxf(v, __shfl_xor(v, 4, 32));
      v = fmaxf(v, __shfl_xor(v, 8, 32));
      rmax[r] = v;
    }
    float esc[8];
#pragma unroll
    for (int r = 0; r < 8; ++r) {
      const float mn = fmaxf(mrow[r], rmax[r]);
      esc[r]  = __expf(mrow[r] - mn);
      mrow[r] = mn;
    }
    float rsum[8];
#pragma unroll
    for (int r = 0; r < 8; ++r) rsum[r] = 0.f;
#pragma unroll
    for (int nt = 0; nt < 4; ++nt)
#pragma unroll
      for (int r = 0; r < 8; ++r) {
        const float p = __expf(s[nt][r] - mrow[r]);
        s[nt][r] = p;
        rsum[r] += p;
      }
#pragma unroll
    for (int r = 0; r < 8; ++r) {
      float v = rsum[r];
      v += __shfl_xor(v, 1, 32);
      v += __shfl_xor(v, 2, 32);
      v += __shfl_xor(v, 4, 32);
      v += __shfl_xor(v, 8, 32);
      lrow[r] = lrow[r] * esc[r] + v;
    }
#pragma unroll
    for (int j = 0; j < 4; ++j)
#pragma unroll
      for (int r = 0; r < 8; ++r) acc[j][r] *= esc[r];

    // P: C-layout -> LDS (per-wave region) -> A-layout fragments
#pragma unroll
    for (int nt = 0; nt < 4; ++nt)
#pragma unroll
      for (int r = 0; r < 8; ++r)
        Plds[w][(r + 8 * hi) * ALD + nt * 16 + lm] = (bf16)s[nt][r];

#pragma unroll
    for (int kc = 0; kc < 2; ++kc) {
      Frag16 pa;
      const bf16* pp = &Plds[w][lm * ALD + kc * 32];
      pa.h[0] = *(const v8bf*)(pp + hi * 8);
      pa.h[1] = *(const v8bf*)(pp + 16 + hi * 8);
#pragma unroll
      for (int dt = 0; dt < 4; ++dt) {
        Frag16 vb;
        const bf16* p = &Vlds[(dt * 16 + lm) * ALD + kc * 32 + hi * 8];
        vb.h[0] = *(const v8bf*)p;
        vb.h[1] = *(const v8bf*)(p + 16);
        acc[dt] = wmma_bf16(pa.v, vb.v, acc[dt]);
      }
    }
    __syncthreads();
  }

  float inv[8];
#pragma unroll
  for (int r = 0; r < 8; ++r) inv[r] = 1.0f / lrow[r];
#pragma unroll
  for (int dt = 0; dt < 4; ++dt) {
    const int col = h * DKH + dt * 16 + lm;
#pragma unroll
    for (int r = 0; r < 8; ++r) {
      const int srow = qblk * 128 + w * 16 + r + 8 * hi;
      ctx[((size_t)(b * SEQ + srow)) * D_MODEL + col] = (bf16)(acc[dt][r] * inv[r]);
    }
  }
}

// ------------------------------------------------------------------ launch ----
extern "C" void kernel_launch(void* const* d_in, const int* in_sizes, int n_in,
                              void* d_out, int out_size, void* d_ws,
                              size_t ws_size, hipStream_t stream) {
  const float* x    = (const float*)d_in[0];
  const int*   mask = (const int*)d_in[1];
  const float* Wq = (const float*)d_in[2];
  const float* bq = (const float*)d_in[3];
  const float* Wk = (const float*)d_in[4];
  const float* bk = (const float*)d_in[5];
  const float* Wv = (const float*)d_in[6];
  const float* bv = (const float*)d_in[7];
  const float* Wo = (const float*)d_in[8];
  const float* bo = (const float*)d_in[9];
  float* out = (float*)d_out;

  char* ws = (char*)d_ws;
  bf16* xb  = (bf16*)(ws + ((size_t)0  << 20));  // 16 MB   x in bf16
  bf16* wqb = (bf16*)(ws + ((size_t)16 << 20));  //  2 MB
  bf16* wkb = (bf16*)(ws + ((size_t)18 << 20));
  bf16* wvb = (bf16*)(ws + ((size_t)20 << 20));
  bf16* wob = (bf16*)(ws + ((size_t)22 << 20));
  bf16* Qb  = (bf16*)(ws + ((size_t)24 << 20));  // 16 MB [B,H,S,DK]
  bf16* Kb  = (bf16*)(ws + ((size_t)40 << 20));  // 16 MB [B,H,S,DK]
  bf16* Vtb = (bf16*)(ws + ((size_t)56 << 20));  // 16 MB [B,H,DK,S]
  bf16* ctx = xb;  // xb is dead after the QKV GEMMs; reuse for ctx [B,S,D]

  const int nx = MTOT * D_MODEL;
  const int nw = D_MODEL * D_MODEL;
  cvt_f32_bf16<<<(nx + 255) / 256, 256, 0, stream>>>(x, xb, nx);
  cvt_f32_bf16<<<(nw + 255) / 256, 256, 0, stream>>>(Wq, wqb, nw);
  cvt_f32_bf16<<<(nw + 255) / 256, 256, 0, stream>>>(Wk, wkb, nw);
  cvt_f32_bf16<<<(nw + 255) / 256, 256, 0, stream>>>(Wv, wvb, nw);
  cvt_f32_bf16<<<(nw + 255) / 256, 256, 0, stream>>>(Wo, wob, nw);

  dim3 gg(D_MODEL / GBN, MTOT / GBM);  // (8, 64)
  gemm_bf16<MODE_QK><<<gg, 256, 0, stream>>>(xb, wqb, bq, Qb);
  gemm_bf16<MODE_QK><<<gg, 256, 0, stream>>>(xb, wkb, bk, Kb);
  gemm_bf16<MODE_VT><<<gg, 256, 0, stream>>>(xb, wvb, bv, Vtb);

  dim3 ga(SEQ / 128, BATCH * HEADS);   // (16, 64)
  attention_fused<<<ga, 256, 0, stream>>>(Qb, Kb, Vtb, mask, ctx);

  gemm_bf16<MODE_F32><<<gg, 256, 0, stream>>>(ctx, wob, bo, out);
}
